// BiLSTMPoolSlow_22686017257707
// MI455X (gfx1250) — matrix operand order repaired and checked by reading, not compiled, not run on hardware
//
#include <hip/hip_runtime.h>
#include <hip/hip_bf16.h>
#include <math.h>

// ---------------- constants (from reference) ----------------
#define BSZ   512
#define TLEN  64
#define HD    256          // hidden
#define G4    1024         // 4H
#define DIN   300          // embedding dim
#define DP    320          // padded K for layer-0 input GEMM (mult of 32)
#define H2    512          // 2H
#define MROWS (TLEN * BSZ) // 32768

// LDS padded strides (bank-conflict avoidance, 64 banks x 4B)
#define GPAD  1040         // gate row stride (floats): 1040 % 64 == 16
#define HBP   264          // hbf row stride (bf16): 528B -> 132 dw -> 4-bank rotation

typedef __bf16 bf16_t;
typedef bf16_t v16bf __attribute__((ext_vector_type(16)));
typedef bf16_t v4bf  __attribute__((ext_vector_type(4)));
typedef float  v8f   __attribute__((ext_vector_type(8)));

union Frag16 { v16bf v; uint4 q[2]; };

__device__ __forceinline__ float4 f4sum3(float4 a, float4 b, float4 c) {
  return make_float4(a.x + b.x + c.x, a.y + b.y + c.y,
                     a.z + b.z + c.z, a.w + b.w + c.w);
}

// ---------------- WMMA bf16 GEMM (input-side, hoisted) ----------------
// C[M,N] (f32) = A[M,K] (bf16, row-major, lda) x Bpk (bf16, fragment-packed)
// Bpk layout: [(n_tile16 * (K/32) + kb) * 32 + lane] * 16 + e   (contiguous 32B/lane)
// Each wave computes a 16x64 strip. Grid is exact (EXEC all-1s for WMMA).
__global__ __launch_bounds__(256)
void gemm_bf16_wmma(const bf16_t* __restrict__ A, int lda,
                    const bf16_t* __restrict__ Bpk,
                    float* __restrict__ C, int ldc,
                    int M, int N, int K) {
  const int wid  = (blockIdx.x * 256 + threadIdx.x) >> 5;
  const int lane = threadIdx.x & 31;
  const int nGrp = N >> 6;
  const int tm   = wid / nGrp;
  const int tg   = wid - tm * nGrp;
  const int half = lane >> 4;
  const int l16  = lane & 15;
  const int m    = (tm << 4) + l16;
  const int nkb  = K >> 5;

  v8f acc[4] = {v8f{}, v8f{}, v8f{}, v8f{}};

  const bf16_t* arow = A + (size_t)m * lda;
  const size_t tileStride = (size_t)nkb * 512;   // elements between 16-col tiles
  for (int kb = 0; kb < nkb; ++kb) {
    Frag16 a;
    const uint4* ap = (const uint4*)(arow + (kb << 5) + (half << 3));
    a.q[0] = ap[0];
    a.q[1] = ap[2];  // +16 bf16 = +32B
    // batch all B-fragment loads, then issue WMMAs (incremental LOADcnt drain)
    Frag16 bf[4];
    const bf16_t* bb0 = Bpk + ((size_t)tg * 4) * tileStride + (size_t)kb * 512 + lane * 16;
#pragma unroll
    for (int i = 0; i < 4; ++i) {
      const uint4* bp = (const uint4*)(bb0 + (size_t)i * tileStride);
      bf[i].q[0] = bp[0];
      bf[i].q[1] = bp[1];
      __builtin_prefetch(bb0 + (size_t)i * tileStride + 512, 0, 3);  // next kb (speculative)
    }
#pragma unroll
    for (int i = 0; i < 4; ++i)
      acc[i] = __builtin_amdgcn_wmma_f32_16x16x32_bf16(
          false, a.v, false, bf[i].v, (short)0, acc[i], false, false);
  }

  const int mrow0 = (tm << 4) + (half << 3);
#pragma unroll
  for (int i = 0; i < 4; ++i) {
    float* cp = C + (size_t)mrow0 * ldc + (tg << 6) + (i << 4) + l16;
#pragma unroll
    for (int r = 0; r < 8; ++r) cp[(size_t)r * ldc] = acc[i][r];
  }
}

// ---------------- fused bidirectional LSTM layer ----------------
// 64 blocks x 256 threads. blockIdx<32: forward cell; else backward cell.
// Block owns batch rows 16*bb..16*bb+15; the recurrence is block-local, so
// all 64 timesteps run inside one kernel with only workgroup barriers.
__device__ __forceinline__ float sigf(float x) { return 1.f / (1.f + __expf(-x)); }

__global__ __launch_bounds__(256)
void lstm_layer_kernel(const float* __restrict__ Gf,    // [T*B,4H] fwd pre-gates
                       const float* __restrict__ Gb,    // [T*B,4H] bwd pre-gates
                       const bf16_t* __restrict__ BhhF, // packed Whh fwd
                       const bf16_t* __restrict__ BhhB, // packed Whh bwd
                       const float* __restrict__ biasF,
                       const float* __restrict__ biasB,
                       const int* __restrict__ lengths,
                       bf16_t* __restrict__ xout,       // layer-1 input [T*B,2H] or null
                       float*  __restrict__ fout) {     // layer-2 output [T*B,2H] or null
  extern __shared__ char smem[];
  float*  sh_g  = (float*)smem;                           // [16][GPAD]  66560 B
  float*  sh_h  = (float*)(smem + 16 * GPAD * 4);         // [16][HD]    16384 B
  float*  sh_c  = sh_h + 16 * HD;                         // [16][HD]    16384 B
  bf16_t* sh_hb = (bf16_t*)(sh_c + 16 * HD);              // [16][HBP]    8448 B
  float*  sh_bi = (float*)(sh_hb + 16 * HBP);             // [G4]         4096 B
  int*    sh_ln = (int*)(sh_bi + G4);                     // [16]           64 B

  const int tid  = threadIdx.x;
  const int dir  = blockIdx.x >> 5;        // 0 fwd, 1 bwd
  const int bb   = blockIdx.x & 31;        // batch tile (16 rows)
  const int w    = tid >> 5;               // wave 0..7
  const int lane = tid & 31;
  const int half = lane >> 4;
  const int l16  = lane & 15;

  const float*  gpre = dir ? Gb : Gf;
  const bf16_t* Bpk  = dir ? BhhB : BhhF;
  const float*  bias = dir ? biasB : biasF;
  const int     coff = dir ? HD : 0;       // output column offset

  // init state + cached constants
  for (int i = tid; i < 16 * HD; i += 256) { sh_h[i] = 0.f; sh_c[i] = 0.f; }
  for (int i = tid; i < 16 * HBP; i += 256) sh_hb[i] = (bf16_t)0.f;
  for (int i = tid; i < G4; i += 256) sh_bi[i] = bias[i];
  if (tid < 16) sh_ln[tid] = lengths[bb * 16 + tid];
  __syncthreads();

  // pointwise-phase indices: thread handles 4 consecutive hidden units
  const int sub = tid >> 6;           // row-within-quad (0..3)
  const int j0  = (tid & 63) << 2;    // hidden column base (0..252)
  const float4 bi_i = *(const float4*)(sh_bi +       j0);
  const float4 bi_f = *(const float4*)(sh_bi + 256 + j0);
  const float4 bi_g = *(const float4*)(sh_bi + 512 + j0);
  const float4 bi_o = *(const float4*)(sh_bi + 768 + j0);

  for (int ti = 0; ti < TLEN; ++ti) {
    const int t = dir ? (TLEN - 1 - ti) : ti;

    // prefetch next timestep's pre-gate rows (speculative; OOB is harmless)
    {
      const int tn = dir ? (t - 1) : (t + 1);
      const char* pf = (const char*)(gpre + (size_t)(tn * BSZ + bb * 16) * G4) + tid * 256;
      __builtin_prefetch(pf, 0, 3);
      __builtin_prefetch(pf + 128, 0, 3);
    }

    // ---- GEMM phase: gates[16][1024] = h[16][256] x Whh^T, via WMMA ----
    v8f acc[8] = {v8f{}, v8f{}, v8f{}, v8f{}, v8f{}, v8f{}, v8f{}, v8f{}};
    const bf16_t* arow = sh_hb + l16 * HBP;   // row = l16 (padded stride)
#pragma unroll
    for (int kb = 0; kb < 8; ++kb) {
      Frag16 a;
      const uint4* ap = (const uint4*)(arow + (kb << 5) + (half << 3));
      a.q[0] = ap[0];          // ds_load_b128
      a.q[1] = ap[2];
      Frag16 bf[8];
      const bf16_t* bb0 = Bpk + (size_t)(w * 8) * 4096 + (size_t)kb * 512 + lane * 16;
#pragma unroll
      for (int i = 0; i < 8; ++i) {
        const uint4* bp = (const uint4*)(bb0 + (size_t)i * 4096);
        bf[i].q[0] = bp[0];
        bf[i].q[1] = bp[1];
      }
#pragma unroll
      for (int i = 0; i < 8; ++i)
        acc[i] = __builtin_amdgcn_wmma_f32_16x16x32_bf16(
            false, a.v, false, bf[i].v, (short)0, acc[i], false, false);
    }
    // store accumulators to padded LDS gate buffer
#pragma unroll
    for (int i = 0; i < 8; ++i) {
      float* gp = sh_g + (size_t)(half << 3) * GPAD + (w << 7) + (i << 4) + l16;
#pragma unroll
      for (int r = 0; r < 8; ++r) gp[(size_t)r * GPAD] = acc[i][r];
    }
    __syncthreads();

    // ---- pointwise phase: 16 rows x 256 hidden, float4-vectorized ----
#pragma unroll
    for (int kq = 0; kq < 4; ++kq) {
      const int k  = kq * 4 + sub;                 // local batch row 0..15
      const int bg = bb * 16 + k;                  // global batch row
      const size_t grow = (size_t)(t * BSZ + bg) * G4;
      const float* gl = sh_g + (size_t)k * GPAD;
      float4 VI = f4sum3(*(const float4*)(gl +       j0),
                         *(const float4*)(gpre + grow +       j0), bi_i);
      float4 VF = f4sum3(*(const float4*)(gl + 256 + j0),
                         *(const float4*)(gpre + grow + 256 + j0), bi_f);
      float4 VG = f4sum3(*(const float4*)(gl + 512 + j0),
                         *(const float4*)(gpre + grow + 512 + j0), bi_g);
      float4 VO = f4sum3(*(const float4*)(gl + 768 + j0),
                         *(const float4*)(gpre + grow + 768 + j0), bi_o);
      float4 CP = *(const float4*)(sh_c + k * HD + j0);
      float4 HP = *(const float4*)(sh_h + k * HD + j0);
      const bool msk = t < sh_ln[k];
      float4 HN, CN, HO;
      const float* vi = (const float*)&VI;
      const float* vf = (const float*)&VF;
      const float* vg = (const float*)&VG;
      const float* vo = (const float*)&VO;
      const float* cp = (const float*)&CP;
      const float* hp = (const float*)&HP;
      float* hN = (float*)&HN;  float* cN = (float*)&CN;  float* ho = (float*)&HO;
#pragma unroll
      for (int c = 0; c < 4; ++c) {
        float i_ = sigf(vi[c]), f_ = sigf(vf[c]), o_ = sigf(vo[c]);
        float g_ = tanhf(vg[c]);
        float cn = f_ * cp[c] + i_ * g_;
        float hn = o_ * tanhf(cn);
        hN[c] = msk ? hn : hp[c];
        cN[c] = msk ? cn : cp[c];
        ho[c] = msk ? hn : 0.f;
      }
      *(float4*)(sh_h + k * HD + j0) = HN;
      *(float4*)(sh_c + k * HD + j0) = CN;
      *(v4bf*)(sh_hb + k * HBP + j0) =
          v4bf{(bf16_t)hN[0], (bf16_t)hN[1], (bf16_t)hN[2], (bf16_t)hN[3]};
      const size_t orow = (size_t)(t * BSZ + bg);
      if (xout)
        *(v4bf*)(xout + orow * H2 + coff + j0) =
            v4bf{(bf16_t)ho[0], (bf16_t)ho[1], (bf16_t)ho[2], (bf16_t)ho[3]};
      if (fout)
        *(float4*)(fout + orow * H2 + coff + j0) = HO;
    }
    __syncthreads();   // protect sh_hb / sh_g for next step
  }
}

// ---------------- weight packing into B-fragment layout ----------------
__global__ void pack_b_kernel(const float* __restrict__ W, int Din, int Kpad,
                              bf16_t* __restrict__ out, int total) {
  int idx = blockIdx.x * blockDim.x + threadIdx.x;
  if (idx >= total) return;
  int e    = idx & 15;
  int lane = (idx >> 4) & 31;
  int rest = idx >> 9;
  int nkb  = Kpad >> 5;
  int kb   = rest % nkb;
  int n0   = rest / nkb;
  int n = (n0 << 4) + (lane & 15);
  int k = (kb << 5) + ((lane >> 4) << 3) + e + ((e >> 3) << 3);
  float v = (k < Din) ? W[(size_t)n * Din + k] : 0.f;
  out[idx] = (bf16_t)v;
}

__global__ void bias_kernel(const float* __restrict__ bi,
                            const float* __restrict__ bh,
                            float* __restrict__ out) {
  int i = blockIdx.x * blockDim.x + threadIdx.x;
  if (i < G4) out[i] = bi[i] + bh[i];
}

// ---------------- embedding gather -> bf16, K padded to 320 ----------------
__global__ void embed_kernel(const int* __restrict__ sents,
                             const float* __restrict__ emb,
                             bf16_t* __restrict__ xbf) {
  int idx = blockIdx.x * blockDim.x + threadIdx.x;
  if (idx >= MROWS * DP) return;
  int r = idx / DP;          // r = t*B + b
  int d = idx - r * DP;
  int t = r / BSZ;
  int b = r - t * BSZ;
  int tok = sents[b * TLEN + t];
  float v = (d < DIN) ? emb[(size_t)tok * DIN + d] : 0.f;
  xbf[idx] = (bf16_t)v;
}

// ---------------- pooling: last / masked-max / masked-mean (float4) ----------------
__global__ void pool_kernel(const float* __restrict__ outF,   // [T*B, 2H]
                            const int* __restrict__ lengths,
                            float* __restrict__ pooled) {     // [B, 6H]
  int idx = blockIdx.x * 256 + threadIdx.x;   // B * 128 quads
  if (idx >= BSZ * (H2 / 4)) return;
  int b  = idx >> 7;
  int c4 = (idx & 127) << 2;
  int len = lengths[b];
  float4 last = *(const float4*)(outF + ((size_t)(len - 1) * BSZ + b) * H2 + c4);
  float4 mx = make_float4(-INFINITY, -INFINITY, -INFINITY, -INFINITY);
  float4 sm = make_float4(0.f, 0.f, 0.f, 0.f);
  for (int t = 0; t < len; ++t) {
    float4 v = *(const float4*)(outF + ((size_t)t * BSZ + b) * H2 + c4);
    mx.x = fmaxf(mx.x, v.x); mx.y = fmaxf(mx.y, v.y);
    mx.z = fmaxf(mx.z, v.z); mx.w = fmaxf(mx.w, v.w);
    sm.x += v.x; sm.y += v.y; sm.z += v.z; sm.w += v.w;
  }
  float inv = 1.f / (float)len;
  sm.x *= inv; sm.y *= inv; sm.z *= inv; sm.w *= inv;
  *(float4*)(pooled + (size_t)b * 1536 +        c4) = last;
  *(float4*)(pooled + (size_t)b * 1536 +  512 + c4) = mx;
  *(float4*)(pooled + (size_t)b * 1536 + 1024 + c4) = sm;
}

// ---------------- final FC: one wave per row, shuffle reduction ----------------
__global__ __launch_bounds__(256)
void fc_kernel(const float* __restrict__ pooled,
               const float* __restrict__ fcw, const float* __restrict__ fcb,
               float* __restrict__ out) {
  int b    = (blockIdx.x * 256 + threadIdx.x) >> 5;
  int lane = threadIdx.x & 31;
  float s = 0.f;
  for (int c = lane; c < 1536; c += 32)
    s += pooled[(size_t)b * 1536 + c] * fcw[c];
#pragma unroll
  for (int off = 16; off > 0; off >>= 1) s += __shfl_xor(s, off, 32);
  if (lane == 0) out[b] = s + fcb[0];
}

// ---------------- host side ----------------
static inline char* bump(char*& p, size_t bytes) {
  char* r = p;
  p += (bytes + 255) & ~(size_t)255;
  return r;
}

extern "C" void kernel_launch(void* const* d_in, const int* in_sizes, int n_in,
                              void* d_out, int out_size, void* d_ws, size_t ws_size,
                              hipStream_t stream) {
  const int*   sents   = (const int*)d_in[0];
  const int*   lengths = (const int*)d_in[1];
  const float* emb     = (const float*)d_in[2];
  const float* w_ih[4] = {(const float*)d_in[3], (const float*)d_in[7],
                          (const float*)d_in[11], (const float*)d_in[15]};
  const float* w_hh[4] = {(const float*)d_in[4], (const float*)d_in[8],
                          (const float*)d_in[12], (const float*)d_in[16]};
  const float* b_ih[4] = {(const float*)d_in[5], (const float*)d_in[9],
                          (const float*)d_in[13], (const float*)d_in[17]};
  const float* b_hh[4] = {(const float*)d_in[6], (const float*)d_in[10],
                          (const float*)d_in[14], (const float*)d_in[18]};
  const float* fcw = (const float*)d_in[19];
  const float* fcb = (const float*)d_in[20];
  float* out = (float*)d_out;

  // workspace carve-out
  char* p = (char*)d_ws;
  bf16_t* xbf   = (bf16_t*)bump(p, (size_t)MROWS * DP * 2);
  bf16_t* x1    = (bf16_t*)bump(p, (size_t)MROWS * H2 * 2);
  float*  Gf    = (float*)bump(p, (size_t)MROWS * G4 * 4);
  float*  Gb    = (float*)bump(p, (size_t)MROWS * G4 * 4);
  float*  outF  = (float*)bump(p, (size_t)MROWS * H2 * 4);
  bf16_t* Bih0f = (bf16_t*)bump(p, (size_t)DP * G4 * 2);
  bf16_t* Bih0b = (bf16_t*)bump(p, (size_t)DP * G4 * 2);
  bf16_t* Bih1f = (bf16_t*)bump(p, (size_t)H2 * G4 * 2);
  bf16_t* Bih1b = (bf16_t*)bump(p, (size_t)H2 * G4 * 2);
  bf16_t* Bhh[4];
  for (int i = 0; i < 4; ++i) Bhh[i] = (bf16_t*)bump(p, (size_t)HD * G4 * 2);
  float* bias[4];
  for (int i = 0; i < 4; ++i) bias[i] = (float*)bump(p, (size_t)G4 * 4);
  float* pooled = (float*)bump(p, (size_t)BSZ * 1536 * 4);
  if ((size_t)(p - (char*)d_ws) > ws_size) return;

  auto gemm = [&](const bf16_t* A, int lda, const bf16_t* Bpk, float* C,
                  int M, int N, int K) {
    int waves = (M / 16) * (N / 64);
    gemm_bf16_wmma<<<waves / 8, 256, 0, stream>>>(A, lda, Bpk, C, N, M, N, K);
  };

  // --- packing ---
  embed_kernel<<<(MROWS * DP + 255) / 256, 256, 0, stream>>>(sents, emb, xbf);
  pack_b_kernel<<<(G4 * DP + 255) / 256, 256, 0, stream>>>(w_ih[0], DIN, DP, Bih0f, G4 * DP);
  pack_b_kernel<<<(G4 * DP + 255) / 256, 256, 0, stream>>>(w_ih[1], DIN, DP, Bih0b, G4 * DP);
  pack_b_kernel<<<(G4 * H2 + 255) / 256, 256, 0, stream>>>(w_ih[2], H2, H2, Bih1f, G4 * H2);
  pack_b_kernel<<<(G4 * H2 + 255) / 256, 256, 0, stream>>>(w_ih[3], H2, H2, Bih1b, G4 * H2);
  for (int i = 0; i < 4; ++i) {
    pack_b_kernel<<<(G4 * HD + 255) / 256, 256, 0, stream>>>(w_hh[i], HD, HD, Bhh[i], G4 * HD);
    bias_kernel<<<4, 256, 0, stream>>>(b_ih[i], b_hh[i], bias[i]);
  }

  const size_t shBytes = (size_t)16 * GPAD * 4   // gates
                       + (size_t)16 * HD * 4 * 2 // h, c
                       + (size_t)16 * HBP * 2    // hbf
                       + (size_t)G4 * 4          // bias
                       + 64;                     // lengths

  // --- layer 0: hoisted input GEMMs + fused bidirectional recurrence ---
  gemm(xbf, DP, Bih0f, Gf, MROWS, G4, DP);
  gemm(xbf, DP, Bih0b, Gb, MROWS, G4, DP);
  lstm_layer_kernel<<<64, 256, shBytes, stream>>>(Gf, Gb, Bhh[0], Bhh[1],
                                                  bias[0], bias[1], lengths,
                                                  x1, nullptr);

  // --- layer 1 ---
  gemm(x1, H2, Bih1f, Gf, MROWS, G4, H2);
  gemm(x1, H2, Bih1b, Gb, MROWS, G4, H2);
  lstm_layer_kernel<<<64, 256, shBytes, stream>>>(Gf, Gb, Bhh[2], Bhh[3],
                                                  bias[2], bias[3], lengths,
                                                  nullptr, outF);

  // --- pooling + FC ---
  pool_kernel<<<(BSZ * H2 / 4 + 255) / 256, 256, 0, stream>>>(outF, lengths, pooled);
  fc_kernel<<<BSZ / 8, 256, 0, stream>>>(pooled, fcw, fcb, out);
}
